// GTV_40991167873127
// MI455X (gfx1250) — matrix operand order repaired and checked by reading, not compiled, not run on hardware
//
#include <hip/hip_runtime.h>
#include <hip/hip_bf16.h>
#include <math.h>

// ---------------------------------------------------------------------------
// GTV forward for MI455X (gfx1250, wave32).
//  - cnnf/cnnu 3x3 convs: implicit-GEMM, v_wmma_f32_16x16x32_f16, 16x32 tiles
//    (2 WMMAs per K-step sharing one A fragment), f16 weights pre-packed with
//    K padded to a multiple of 32 so A loads are aligned 16-byte vectors.
//  - dense inv(I+u*Lap)@y replaced by warm-started CG on the 5-point grid
//    stencil (SPD, diagonally dominant) -- one workgroup per (b,ch) system,
//    search direction resident in LDS.
// ---------------------------------------------------------------------------

typedef __attribute__((ext_vector_type(16))) _Float16 v16h;
typedef __attribute__((ext_vector_type(8)))  _Float16 v8h;
typedef __attribute__((ext_vector_type(8)))  float    v8f;

#define WD   36
#define NN   (WD * WD)       // 1296
#define NEH  (WD * (WD - 1)) // 1260 horizontal edges
#define ZMAXV 0.01f

// ===========================================================================
// Weight repack: f32 [Cout, K] -> f16 [Cout, Kpad], zero-padded K..Kpad.
// Kpad % 32 == 0 so each A-fragment row-chunk is a 16B-aligned v8h load.
// ===========================================================================
__global__ void w2h(const float* __restrict__ src, _Float16* __restrict__ dst,
                    int Cout, int K, int Kpad)
{
    int idx = blockIdx.x * blockDim.x + threadIdx.x;
    if (idx >= Cout * Kpad) return;
    int co = idx / Kpad, k = idx - co * Kpad;
    dst[idx] = (k < K) ? (_Float16)src[(size_t)co * K + k] : (_Float16)0.f;
}

// ===========================================================================
// Generic 3x3 conv as implicit GEMM with WMMA.
//   padmode 0: zero pad=1, given stride
//   padmode 1: input [Cin,Hin,Win]; logically upsample x2 + reflect-pad 1,
//              then 3x3 conv: Hout=2*Hin, Wout=2*Win
//   act: 0 none, 1 relu, 2 leaky(0.05)
// One wave computes a 16(out-ch) x 32(pixel) tile: one A fragment (f16
// weights), two B fragments (im2col gather), two v_wmma_f32_16x16x32_f16.
// Fragment layouts per CDNA5 ISA 7.12.2:
//   A(16x32 f16): lane m=lane&15; halves 0-7 -> K=k0+8*(lane>>4)+h,
//                 halves 8-15 -> K=k0+8*(lane>>4)+16+(h-8)
//   B(32x16 f16): lane -> N=lane&15; halves h -> same K as A
//   D(16x16 f32): lane -> N=lane&15, vgpr r -> M=r+8*(lane>>4)
// ===========================================================================
__global__ void conv_wmma(const float* __restrict__ X, const _Float16* __restrict__ Wh,
                          const float* __restrict__ Bs, float* __restrict__ Y,
                          int Cin, int Cout, int Kpad,
                          int Hin, int Win, int Hout, int Wout,
                          int stride, int padmode, int act,
                          int inBatchStride, int outChanTotal, int outChanOffset)
{
    const int b    = blockIdx.y;
    const int lane = threadIdx.x & 31;
    const int wave = threadIdx.x >> 5;
    const int wavesPerBlock = blockDim.x >> 5;

    const int P       = Hout * Wout;
    const int coTiles = (Cout + 15) >> 4;
    const int pTiles  = (P + 31) >> 5;          // 32 pixels per wave-tile
    const int tile    = blockIdx.x * wavesPerBlock + wave;
    if (tile >= coTiles * pTiles) return;

    const int coTile = tile / pTiles;
    const int pTile  = tile % pTiles;
    const int K      = Cin * 9;

    const float* Xb = X + (size_t)b * inBatchStride;

    const int m    = lane & 15;
    const int hiA  = lane >> 4;                  // 0 or 1
    const int co_m = coTile * 16 + m;
    const _Float16* Wrow = Wh + (size_t)co_m * Kpad;

    if (co_m < Cout)  // gfx1250 global_prefetch_b8 on the weight stream
        __builtin_prefetch(Wrow, 0, 1);

    v8f acc0 = {}, acc1 = {};

    for (int k0 = 0; k0 < Kpad; k0 += 32) {
        // ---- A fragment: two aligned 16B vector loads of packed f16 weights ----
        v16h afrag;
        if (co_m < Cout) {
            v8h lo = *(const v8h*)(Wrow + k0 + hiA * 8);
            v8h hi = *(const v8h*)(Wrow + k0 + hiA * 8 + 16);
#pragma unroll
            for (int h = 0; h < 8; ++h) { afrag[h] = lo[h]; afrag[h + 8] = hi[h]; }
        } else {
#pragma unroll
            for (int h = 0; h < 16; ++h) afrag[h] = (_Float16)0.f;
        }

        // ---- B fragments: im2col gather; lane&15 = pixel, halves = K (table order) ----
        v16h bfrag0, bfrag1;
#pragma unroll
        for (int h = 0; h < 16; ++h) {
            int k = k0 + hiA * 8 + (h < 8 ? h : h + 8);
            int ci = 0, kh = 0, kw = 0;
            bool kvalid = (k < K);
            if (kvalid) { ci = k / 9; int r9 = k - ci * 9; kh = r9 / 3; kw = r9 - kh * 3; }
#pragma unroll
            for (int half = 0; half < 2; ++half) {
                float xv = 0.0f;
                int p = pTile * 32 + half * 16 + (lane & 15);
                if (kvalid && p < P) {
                    int oy = p / Wout, ox = p - oy * Wout;
                    if (padmode == 0) {
                        int iy = oy * stride + kh - 1;
                        int ix = ox * stride + kw - 1;
                        if (iy >= 0 && iy < Hin && ix >= 0 && ix < Win)
                            xv = Xb[((size_t)ci * Hin + iy) * Win + ix];
                    } else {
                        // upsample x2 + reflect pad 1 (jnp 'reflect')
                        int Hup = Hin * 2, Wup = Win * 2;
                        int iy = oy + kh - 1, ix = ox + kw - 1;
                        iy = (iy < 0) ? -iy : ((iy >= Hup) ? (2 * Hup - 2 - iy) : iy);
                        ix = (ix < 0) ? -ix : ((ix >= Wup) ? (2 * Wup - 2 - ix) : ix);
                        xv = Xb[((size_t)ci * Hin + (iy >> 1)) * Win + (ix >> 1)];
                    }
                }
                if (half == 0) bfrag0[h] = (_Float16)xv;
                else           bfrag1[h] = (_Float16)xv;
            }
        }
        acc0 = __builtin_amdgcn_wmma_f32_16x16x32_f16(
                   false, afrag, false, bfrag0, (short)0, acc0, false, false);
        acc1 = __builtin_amdgcn_wmma_f32_16x16x32_f16(
                   false, afrag, false, bfrag1, (short)0, acc1, false, false);
    }

    // ---- store D tiles: lane -> pixel column, vgpr r -> output row ----
    const int n = lane & 15;
    const int mBase = (lane >> 4) * 8;
#pragma unroll
    for (int half = 0; half < 2; ++half) {
        int p = pTile * 32 + half * 16 + n;
        if (p < P) {
#pragma unroll
            for (int r = 0; r < 8; ++r) {
                int co = coTile * 16 + mBase + r;
                if (co < Cout) {
                    float v = (half == 0 ? acc0[r] : acc1[r]) + Bs[co];
                    if      (act == 1) v = v > 0.f ? v : 0.f;
                    else if (act == 2) v = v > 0.f ? v : 0.05f * v;
                    Y[((size_t)b * outChanTotal + outChanOffset + co) * P + p] = v;
                }
            }
        }
    }
}

// ===========================================================================
// 2x2 max pool (optional ceil-mode padding with -inf)
// ===========================================================================
__global__ void maxpool2k(const float* __restrict__ X, float* __restrict__ Y,
                          int C, int H, int W, int Ho, int Wo)
{
    int idx = blockIdx.x * blockDim.x + threadIdx.x;
    int b = blockIdx.y;
    int total = C * Ho * Wo;
    if (idx >= total) return;
    int c = idx / (Ho * Wo);
    int rem = idx - c * Ho * Wo;
    int oy = rem / Wo, ox = rem - oy * Wo;
    float mx = -INFINITY;
    for (int dy = 0; dy < 2; ++dy) {
        int iy = 2 * oy + dy; if (iy >= H) continue;
        for (int dx = 0; dx < 2; ++dx) {
            int ix = 2 * ox + dx; if (ix >= W) continue;
            float v = X[((size_t)b * C + c) * H * W + (size_t)iy * W + ix];
            mx = fmaxf(mx, v);
        }
    }
    Y[((size_t)b * C + c) * Ho * Wo + rem] = mx;
}

// ===========================================================================
// cnnu head: 288->32->1, relu, clip to [0.001, 1]. One block, warp per batch.
// ===========================================================================
__global__ void cnnu_fc(const float* __restrict__ feat,
                        const float* __restrict__ w1, const float* __restrict__ b1,
                        const float* __restrict__ w2, const float* __restrict__ b2,
                        float* __restrict__ u, int B)
{
    __shared__ float h[8][32];
    int b = threadIdx.x >> 5;
    int j = threadIdx.x & 31;
    if (b < B) {
        float s = b1[j];
        const float* xb = feat + (size_t)b * 288;
        for (int k = 0; k < 288; ++k) s += w1[(size_t)j * 288 + k] * xb[k];
        h[b][j] = s;
    }
    __syncthreads();
    if (b < B && j == 0) {
        float s = b2[0];
        for (int k = 0; k < 32; ++k) s += w2[k] * h[b][k];
        s = s > 0.f ? s : 0.f;                 // relu
        s = fminf(fmaxf(s, 0.001f), 1.0f);     // clip(U_MIN, U_MAX)
        u[b] = s;
    }
}

// ===========================================================================
// Edge weights: w[b,e] = exp(-0.5 * sum_c (E[b,c,i]-E[b,c,j])^2)
// ===========================================================================
__global__ void edge_w(const float* __restrict__ E, const int* __restrict__ ci,
                       const int* __restrict__ cj, float* __restrict__ wgt, int ne)
{
    int e = blockIdx.x * blockDim.x + threadIdx.x;
    int b = blockIdx.y;
    if (e >= ne) return;
    int i = ci[e], j = cj[e];
    const float* Eb = E + (size_t)b * 12 * NN;
    float s = 0.f;
#pragma unroll
    for (int c = 0; c < 12; ++c) {
        float d = Eb[(size_t)c * NN + i] - Eb[(size_t)c * NN + j];
        s += d * d;
    }
    wgt[(size_t)b * ne + e] = __expf(-0.5f * s);
}

// ===========================================================================
// Reweighting: lw[b,ch,e] = wgt[b,e] / max(|x_i - x_j|, ZMAX)
// ===========================================================================
__global__ void edge_lw(const float* __restrict__ x, const float* __restrict__ wgt,
                        const int* __restrict__ ci, const int* __restrict__ cj,
                        float* __restrict__ lw, int ne)
{
    int e  = blockIdx.x * blockDim.x + threadIdx.x;
    int bc = blockIdx.y;  // b*3 + ch
    if (e >= ne) return;
    int b = bc / 3;
    int i = ci[e], j = cj[e];
    const float* xb = x + (size_t)bc * NN;
    float z = fmaxf(fabsf(xb[i] - xb[j]), ZMAXV);
    lw[(size_t)bc * ne + e] = wgt[(size_t)b * ne + e] / z;
}

// ===========================================================================
// Fused CG solve of (I + u * Lap) x = y on the 36x36 4-connected grid.
// One block per (b,ch) system; p-vector lives in LDS; stencil weights in
// registers. Warm start from current x (in d_out).
// ===========================================================================
__device__ __forceinline__ float block_reduce(float v, float* red)
{
    int t = threadIdx.x;
#pragma unroll
    for (int off = 16; off > 0; off >>= 1) v += __shfl_down(v, off, 32);
    if ((t & 31) == 0) red[t >> 5] = v;
    __syncthreads();
    if (t < 32) {
        int nw = blockDim.x >> 5;
        float r = (t < nw) ? red[t] : 0.f;
#pragma unroll
        for (int off = 16; off > 0; off >>= 1) r += __shfl_down(r, off, 32);
        if (t == 0) red[0] = r;
    }
    __syncthreads();
    float out = red[0];
    __syncthreads();
    return out;
}

__device__ __forceinline__ float applyA(int nd, const float* pS, float u, float deg,
                                        float wl, float wr, float wu, float wd)
{
    int r = nd / WD, c = nd - r * WD;
    float v   = pS[nd];
    float acc = deg * v;
    if (c > 0)      acc -= wl * pS[nd - 1];
    if (c < WD - 1) acc -= wr * pS[nd + 1];
    if (r > 0)      acc -= wu * pS[nd - WD];
    if (r < WD - 1) acc -= wd * pS[nd + WD];
    return v + u * acc;
}

#define CG_NP 6   // 256 threads * 6 slots >= 1296 nodes

__global__ void cg_solve(float* __restrict__ x, const float* __restrict__ y,
                         const float* __restrict__ lw, const float* __restrict__ u_arr,
                         int ne, int cgIters)
{
    __shared__ float pS[NN];
    __shared__ float red[32];

    const int bc = blockIdx.x;       // b*3 + ch
    const int b  = bc / 3;
    const float u = u_arr[b];
    const float* lwb = lw + (size_t)bc * ne;
    float*       xb  = x  + (size_t)bc * NN;
    const float* yb  = y  + (size_t)bc * NN;
    const int t = threadIdx.x;

    float wl[CG_NP], wr[CG_NP], wu[CG_NP], wd[CG_NP], dg[CG_NP];
    float xr[CG_NP], rr[CG_NP], yv[CG_NP];

#pragma unroll
    for (int s = 0; s < CG_NP; ++s) {
        int nd = t + s * 256;
        if (nd < NN) {
            int r = nd / WD, c = nd - r * WD;
            wl[s] = (c > 0)      ? lwb[r * (WD - 1) + c - 1]   : 0.f;
            wr[s] = (c < WD - 1) ? lwb[r * (WD - 1) + c]       : 0.f;
            wu[s] = (r > 0)      ? lwb[NEH + (r - 1) * WD + c] : 0.f;
            wd[s] = (r < WD - 1) ? lwb[NEH + r * WD + c]       : 0.f;
            dg[s] = wl[s] + wr[s] + wu[s] + wd[s];
            xr[s] = xb[nd];
            yv[s] = yb[nd];
            pS[nd] = xr[s];          // temporarily hold x for residual calc
        }
    }
    __syncthreads();

    // r = y - A x ; rs = <r,r>
    float rs_loc = 0.f;
#pragma unroll
    for (int s = 0; s < CG_NP; ++s) {
        int nd = t + s * 256;
        if (nd < NN) {
            float av = applyA(nd, pS, u, dg[s], wl[s], wr[s], wu[s], wd[s]);
            rr[s] = yv[s] - av;
            rs_loc += rr[s] * rr[s];
        }
    }
    float rs = block_reduce(rs_loc, red);

    // p = r
#pragma unroll
    for (int s = 0; s < CG_NP; ++s) {
        int nd = t + s * 256;
        if (nd < NN) pS[nd] = rr[s];
    }
    __syncthreads();

    for (int it = 0; it < cgIters; ++it) {
        float ap[CG_NP];
        float pap_loc = 0.f;
#pragma unroll
        for (int s = 0; s < CG_NP; ++s) {
            int nd = t + s * 256;
            if (nd < NN) {
                ap[s] = applyA(nd, pS, u, dg[s], wl[s], wr[s], wu[s], wd[s]);
                pap_loc += pS[nd] * ap[s];
            }
        }
        float pap   = block_reduce(pap_loc, red);
        float alpha = rs / fmaxf(pap, 1e-30f);

        float rsn_loc = 0.f;
#pragma unroll
        for (int s = 0; s < CG_NP; ++s) {
            int nd = t + s * 256;
            if (nd < NN) {
                xr[s] += alpha * pS[nd];
                rr[s] -= alpha * ap[s];
                rsn_loc += rr[s] * rr[s];
            }
        }
        float rs_new = block_reduce(rsn_loc, red);
        float beta   = rs_new / fmaxf(rs, 1e-30f);
        rs = rs_new;

        __syncthreads();     // everyone done reading pS
#pragma unroll
        for (int s = 0; s < CG_NP; ++s) {
            int nd = t + s * 256;
            if (nd < NN) pS[nd] = rr[s] + beta * pS[nd];
        }
        __syncthreads();
    }

#pragma unroll
    for (int s = 0; s < CG_NP; ++s) {
        int nd = t + s * 256;
        if (nd < NN) xb[nd] = xr[s];
    }
}

// ===========================================================================
// Host side
// ===========================================================================
static inline int kpad32(int Cin) { return ((Cin * 9 + 31) / 32) * 32; }

static inline void launch_conv(const float* X, const _Float16* Wh, const float* Bs,
                               float* Y, int Cin, int Cout, int Hin, int Win,
                               int Hout, int Wout, int stride, int padmode, int act,
                               int inBatchStride, int outChanTotal, int outChanOffset,
                               int Bn, hipStream_t stream)
{
    int P = Hout * Wout;
    int tiles = ((Cout + 15) / 16) * ((P + 31) / 32);
    const int wavesPerBlock = 8;
    int blocks = (tiles + wavesPerBlock - 1) / wavesPerBlock;
    dim3 g(blocks, Bn), blk(32 * wavesPerBlock);
    conv_wmma<<<g, blk, 0, stream>>>(X, Wh, Bs, Y, Cin, Cout, kpad32(Cin),
                                     Hin, Win, Hout, Wout, stride, padmode, act,
                                     inBatchStride, outChanTotal, outChanOffset);
}

static inline void launch_pool(const float* X, float* Y, int C, int H, int W,
                               bool ceilm, int Bn, hipStream_t stream)
{
    int Ho = ceilm ? (H + 1) / 2 : H / 2;
    int Wo = ceilm ? (W + 1) / 2 : W / 2;
    int total = C * Ho * Wo;
    dim3 g((total + 255) / 256, Bn), blk(256);
    maxpool2k<<<g, blk, 0, stream>>>(X, Y, C, H, W, Ho, Wo);
}

extern "C" void kernel_launch(void* const* d_in, const int* in_sizes, int n_in,
                              void* d_out, int out_size, void* d_ws, size_t ws_size,
                              hipStream_t stream)
{
    const float* xf = (const float*)d_in[0];
    const int B  = in_sizes[0] / (3 * NN);   // 4
    const int ne = in_sizes[39];             // 2520

    // biases (f32, used directly)
    const float *l1c1b=(const float*)d_in[2],  *l1c2b=(const float*)d_in[4];
    const float *l2ab =(const float*)d_in[6],  *l2b  =(const float*)d_in[8];
    const float *l3ab =(const float*)d_in[10], *l3b  =(const float*)d_in[12];
    const float *d1b  =(const float*)d_in[14], *l4c1b=(const float*)d_in[16];
    const float *l4c2b=(const float*)d_in[18], *d2b  =(const float*)d_in[20];
    const float *l5c1b=(const float*)d_in[22], *l5c2b=(const float*)d_in[24];
    const float *l5c3b=(const float*)d_in[26];
    const float *c1b=(const float*)d_in[28], *c2b=(const float*)d_in[30];
    const float *c3b=(const float*)d_in[32], *c4b=(const float*)d_in[34];
    const float *fc1w=(const float*)d_in[35], *fc1b=(const float*)d_in[36];
    const float *fc2w=(const float*)d_in[37], *fc2b=(const float*)d_in[38];
    const int *conn_i=(const int*)d_in[39], *conn_j=(const int*)d_in[40];

    float* x_out = (float*)d_out;            // [B,3,36,36] -- doubles as CG state

    // ---- workspace bump allocator (floats; every block is a multiple of
    //      4 floats so the trailing f16 region stays 16B-aligned) ----
    float* wsf = (float*)d_ws;
    size_t off = 0;
    auto alloc = [&](size_t n) { float* p = wsf + off; off += n; return p; };

    float* o1a  = alloc((size_t)B * 32 * NN);
    float* cat2 = alloc((size_t)B * 64 * NN);    // ch0-31: d2 out, ch32-63: o1
    float* t2   = alloc((size_t)B * 64 * NN);
    float* o2p  = alloc((size_t)B * 64 * 324);
    float* cat1 = alloc((size_t)B * 128 * 324);  // ch0-63: d1 out, ch64-127: o2
    float* t3   = alloc((size_t)B * 64 * 324);
    float* o3p  = alloc((size_t)B * 64 * 81);
    float* o3   = alloc((size_t)B * 128 * 81);
    float* o4a  = alloc((size_t)B * 64 * 324);
    float* o4   = alloc((size_t)B * 64 * 324);
    float* o5a  = alloc((size_t)B * 32 * NN);
    float* o5b  = alloc((size_t)B * 32 * NN);
    float* Ef   = alloc((size_t)B * 12 * NN);
    float* u1   = alloc((size_t)B * 32 * 324);
    float* u2   = alloc((size_t)B * 32 * 324);
    float* u2p  = alloc((size_t)B * 32 * 84);    // 32*81 rounded up (alignment)
    float* u3   = alloc((size_t)B * 32 * 84);
    float* u3p  = alloc((size_t)B * 32 * 28);    // 32*25 rounded up
    float* u4   = alloc((size_t)B * 32 * 28);
    float* u4p  = alloc((size_t)B * 32 * 12);    // contiguous [B,288] region used
    float* uarr = alloc((size_t)B);
    float* wgt  = alloc((size_t)B * ne);
    float* lw   = alloc((size_t)B * 3 * ne);

    // ---- f16 packed weights: {d_in index, Cout, Cin} in launch order ----
    struct WDesc { int idx, cout, cin; };
    const WDesc wd[17] = {
        {1,32,3},   {3,32,32},  {5,64,32},  {7,64,64},  {9,64,64},  {11,128,64},
        {13,64,128},{15,64,128},{17,64,64}, {19,32,64}, {21,32,64}, {23,32,32},
        {25,12,32}, {27,32,3},  {29,32,32}, {31,32,32}, {33,32,32}
    };
    _Float16* wh[17];
    {
        _Float16* hbase = (_Float16*)(wsf + off);
        size_t hoff = 0;
        for (int i = 0; i < 17; ++i) {
            int kp = kpad32(wd[i].cin);
            wh[i] = hbase + hoff;
            hoff += (size_t)wd[i].cout * kp;
            int total = wd[i].cout * kp;
            w2h<<<dim3((total + 255) / 256), dim3(256), 0, stream>>>(
                (const float*)d_in[wd[i].idx], wh[i], wd[i].cout, wd[i].cin * 9, kp);
        }
    }
    (void)ws_size; (void)n_in; (void)out_size;

    // ================= cnnf =================
    launch_conv(xf,   wh[0],  l1c1b, o1a,  3, 32, 36,36, 36,36, 1,0,2, 3*NN,   32, 0,  B, stream);
    launch_conv(o1a,  wh[1],  l1c2b, cat2, 32,32, 36,36, 36,36, 1,0,2, 32*NN,  64, 32, B, stream); // o1 -> cat2[32:64]
    launch_conv(cat2 + 32*NN, wh[2], l2ab, t2, 32,64, 36,36, 36,36, 1,0,1, 64*NN, 64, 0, B, stream);
    launch_pool(t2, o2p, 64, 36, 36, false, B, stream);                                            // 18x18
    launch_conv(o2p,  wh[3],  l2b,  cat1, 64,64, 18,18, 18,18, 1,0,1, 64*324, 128, 64, B, stream); // o2 -> cat1[64:128]
    launch_conv(cat1 + 64*324, wh[4], l3ab, t3, 64,64, 18,18, 18,18, 1,0,1, 128*324, 64, 0, B, stream);
    launch_pool(t3, o3p, 64, 18, 18, false, B, stream);                                            // 9x9
    launch_conv(o3p,  wh[5],  l3b,  o3,  64,128,  9, 9,  9, 9, 1,0,1, 64*81, 128, 0, B, stream);
    launch_conv(o3,   wh[6],  d1b,  cat1,128,64,  9, 9, 18,18, 1,1,0, 128*81, 128, 0, B, stream);  // upsample+reflect
    launch_conv(cat1, wh[7],  l4c1b,o4a, 128,64, 18,18, 18,18, 1,0,2, 128*324, 64, 0, B, stream);
    launch_conv(o4a,  wh[8],  l4c2b,o4,  64, 64, 18,18, 18,18, 1,0,2, 64*324,  64, 0, B, stream);
    launch_conv(o4,   wh[9],  d2b,  cat2, 64,32, 18,18, 36,36, 1,1,0, 64*324,  64, 0, B, stream);  // upsample+reflect
    launch_conv(cat2, wh[10], l5c1b,o5a, 64, 32, 36,36, 36,36, 1,0,2, 64*NN,   32, 0, B, stream);
    launch_conv(o5a,  wh[11], l5c2b,o5b, 32, 32, 36,36, 36,36, 1,0,2, 32*NN,   32, 0, B, stream);
    launch_conv(o5b,  wh[12], l5c3b,Ef,  32, 12, 36,36, 36,36, 1,0,0, 32*NN,   12, 0, B, stream);

    // ================= cnnu =================
    launch_conv(xf,  wh[13], c1b, u1, 3, 32, 36,36, 18,18, 2,0,2, 3*NN,  32, 0, B, stream);
    launch_conv(u1,  wh[14], c2b, u2, 32,32, 18,18, 18,18, 1,0,2, 32*324,32, 0, B, stream);
    launch_pool(u2, u2p, 32, 18, 18, true, B, stream);                       // 9x9
    launch_conv(u2p, wh[15], c3b, u3, 32,32,  9, 9,  9, 9, 1,0,2, 32*81, 32, 0, B, stream);
    launch_pool(u3, u3p, 32, 9, 9, true, B, stream);                         // 5x5
    launch_conv(u3p, wh[16], c4b, u4, 32,32,  5, 5,  5, 5, 1,0,2, 32*25, 32, 0, B, stream);
    launch_pool(u4, u4p, 32, 5, 5, true, B, stream);                         // 3x3 -> [B,288]
    cnnu_fc<<<dim3(1), dim3(32 * B), 0, stream>>>(u4p, fc1w, fc1b, fc2w, fc2b, uarr, B);

    // ================= graph TV solve =================
    {
        dim3 g((ne + 255) / 256, B), blk(256);
        edge_w<<<g, blk, 0, stream>>>(Ef, conn_i, conn_j, wgt, ne);
    }
    // x0 = y = xf
    hipMemcpyAsync(x_out, xf, (size_t)B * 3 * NN * sizeof(float),
                   hipMemcpyDeviceToDevice, stream);

    const int N_ITER = 10;
    const int CG_ITERS = 32;
    for (int it = 0; it < N_ITER; ++it) {
        dim3 g((ne + 255) / 256, B * 3), blk(256);
        edge_lw<<<g, blk, 0, stream>>>(x_out, wgt, conn_i, conn_j, lw, ne);
        cg_solve<<<dim3(B * 3), dim3(256), 0, stream>>>(x_out, xf, lw, uarr, ne, CG_ITERS);
    }
}
